// SparseBlock_18554258719214
// MI455X (gfx1250) — matrix operand order repaired and checked by reading, not compiled, not run on hardware
//
#include <hip/hip_runtime.h>
#include <hip/hip_bf16.h>

// ---------------------------------------------------------------------------
// Sparse 3x3x3 conv residual block on MI455X (gfx1250, wave32, WMMA).
//
//   0) memset per-channel sum/sumsq accumulators
//   1) feats f32 -> f16                       (A-operand feed for conv1)
//   2) W1,W2 -> f16 transposed [k][cout][cin] (B-operand feed)
//   3) conv1: async-fill W into LDS (global_load_async_to_lds_b128),
//      WMMA gather-GEMM -> y (f32) + per-channel sum/sumsq
//   4) bn_stats1: fold gamma/beta/mu/rsig into scale/shift
//   5) bn+relu -> h (f16, reuses feats_h buffer)
//   6) conv2: same WMMA kernel on h -> y + sum/sumsq (second set)
//   7) bn_stats2
//   8) bn2 + residual + relu -> d_out
// ---------------------------------------------------------------------------

typedef _Float16 v8h  __attribute__((ext_vector_type(8)));
typedef _Float16 v16h __attribute__((ext_vector_type(16)));
typedef float    v8f  __attribute__((ext_vector_type(8)));

#define NVOX   100000
#define KOFF   27
#define CCH    64
#define NTILES (NVOX / 16)      // 6250, N is a multiple of 16
#define WROWS  (KOFF * CCH)     // 1728 weight rows of 64 halves
#define WPAD   72               // padded row stride in halves (bank-conflict-free)
#define LDSBYTES (WROWS * WPAD * 2)   // 248,832 B  (< 320 KB per WGP)

// Build a v16h WMMA operand from two 16-byte (8-half) aligned chunks.
__device__ __forceinline__ v16h ld16(const _Float16* plo, const _Float16* phi) {
  v8h lo = *(const v8h*)plo;
  v8h hi = *(const v8h*)phi;
  v16h r;
#pragma unroll
  for (int i = 0; i < 8; ++i) { r[i] = lo[i]; r[i + 8] = hi[i]; }
  return r;
}

// gfx1250 async copy: 16 B global -> LDS, tracked on ASYNCcnt (no VGPR bounce).
__device__ __forceinline__ void async_g2l_b128(unsigned lds_byte_off, const void* gaddr) {
  asm volatile("global_load_async_to_lds_b128 %0, %1, off"
               :: "v"(lds_byte_off), "v"(gaddr)
               : "memory");
}
__device__ __forceinline__ void wait_asynccnt0() {
  asm volatile("s_wait_asynccnt 0" ::: "memory");
}

// ----------------------------- prep kernels --------------------------------

__global__ __launch_bounds__(256) void cvt_f32_to_f16(const float* __restrict__ x,
                                                      _Float16* __restrict__ o, int n) {
  int i = blockIdx.x * 256 + threadIdx.x;
  if (i < n) o[i] = (_Float16)x[i];
}

// W[k][cin][cout] (f32) -> Wt[k][cout][cin] (f16), cin contiguous per lane.
__global__ __launch_bounds__(256) void transpose_w(const float* __restrict__ W,
                                                   _Float16* __restrict__ Wt, int total) {
  int i = blockIdx.x * 256 + threadIdx.x;
  if (i >= total) return;
  int kk = i >> 12;           // / (64*64)
  int r  = i & 4095;
  int cout = r >> 6;
  int cin  = r & 63;
  Wt[i] = (_Float16)W[(kk << 12) + (cin << 6) + cout];
}

// ------------------------------ conv kernel --------------------------------
// One wave = one 16-voxel x 64-channel output tile.
// All 27*64*64 f16 weights cached in LDS (async fill, padded rows), reused by
// all 8 waves. 27 offsets x 2 K-chunks x 4 N-blocks = 216 WMMAs per tile.
__global__ __launch_bounds__(256) void conv_wmma(
    const _Float16* __restrict__ featsH,   // [N][64] f16
    const int*      __restrict__ nbr,      // [N][27]
    const _Float16* __restrict__ Wt,       // [27][64][64] f16 (cout-major)
    const float*    __restrict__ bias,     // [64]
    float*          __restrict__ y,        // [N][64] f32 (pre-BN)
    float*          __restrict__ chSum,    // [64]
    float*          __restrict__ chSumSq)  // [64]
{
  extern __shared__ _Float16 Wlds[];       // [1728][72] halves, rows padded

  // ---- cooperative async fill: Wt (global) -> Wlds (LDS), all 256 threads --
  for (int row = threadIdx.x; row < WROWS; row += 256) {
    const unsigned lbase = (unsigned)row * (WPAD * 2);     // 144 B padded rows
    const char*    gbase = (const char*)Wt + (size_t)row * (CCH * 2);
#pragma unroll
    for (int c = 0; c < 8; ++c)
      async_g2l_b128(lbase + c * 16, gbase + c * 16);
  }
  wait_asynccnt0();
  __syncthreads();

  const int wave = threadIdx.x >> 5;
  const int lane = threadIdx.x & 31;
  const int tile = blockIdx.x * 8 + wave;
  if (tile < NTILES) {                     // wave-uniform: EXEC stays all-ones
    const int m  = lane & 15;              // A row, B/C column
    const int kh = lane >> 4;              // K-half select (ISA lane split)
    const int* nrow = nbr + (size_t)(tile * 16 + m) * KOFF;
    // Per-lane B base inside LDS: row (kk*64 + nb*16 + m), halves kh*16 + ...
    const _Float16* wlane = Wlds + (size_t)m * WPAD + kh * 16;

    v8f acc[4];
#pragma unroll
    for (int nb = 0; nb < 4; ++nb)
#pragma unroll
      for (int r = 0; r < 8; ++r) acc[nb][r] = 0.0f;

    for (int kk = 0; kk < KOFF; ++kk) {
      const int idx = nrow[kk];                             // per-lane gather
      const _Float16* fp = featsH + (size_t)idx * CCH;
      // A chunk0: cin 0..31  (halves 0-15 <- cin kh*16+0..15)
      v16h a0 = ld16(fp + kh * 16,      fp + 8 + kh * 16);
      // A chunk1: cin 32..63
      v16h a1 = ld16(fp + 32 + kh * 16, fp + 40 + kh * 16);

      const _Float16* wk = wlane + (size_t)kk * CCH * WPAD;
#pragma unroll
      for (int nb = 0; nb < 4; ++nb) {
        const _Float16* bp = wk + nb * 16 * WPAD;
        v16h b0 = ld16(bp,      bp + 8);      // cin kh*16 + 0..15
        v16h b1 = ld16(bp + 32, bp + 40);     // cin 32 + kh*16 + 0..15
        acc[nb] = __builtin_amdgcn_wmma_f32_16x16x32_f16(
            false, a0, false, b0, (short)0, acc[nb], false, false);
        acc[nb] = __builtin_amdgcn_wmma_f32_16x16x32_f16(
            false, a1, false, b1, (short)0, acc[nb], false, false);
      }
    }

    // Bias add, store raw conv output, accumulate per-channel BN statistics.
#pragma unroll
    for (int nb = 0; nb < 4; ++nb) {
      const int ch = nb * 16 + m;
      const float bv = bias[ch];
      float s = 0.0f, ss = 0.0f;
#pragma unroll
      for (int r = 0; r < 8; ++r) {
        const float val = acc[nb][r] + bv;
        const int row = r + kh * 8;                        // C/D row layout
        y[(size_t)(tile * 16 + row) * CCH + ch] = val;
        s += val; ss += val * val;
      }
      s  += __shfl_xor(s,  16, 32);                        // merge row halves
      ss += __shfl_xor(ss, 16, 32);
      if (kh == 0) {
        atomicAdd(&chSum[ch], s);
        atomicAdd(&chSumSq[ch], ss);
      }
    }
  }
}

// --------------------------- BN finalize / apply ---------------------------

__global__ void bn_stats(const float* __restrict__ sum, const float* __restrict__ sumsq,
                         const float* __restrict__ gamma, const float* __restrict__ beta,
                         float* __restrict__ scale, float* __restrict__ shift) {
  const int c = threadIdx.x;            // 64 threads
  const float inv = 1.0f / (float)NVOX;
  const float mu  = sum[c] * inv;
  const float var = sumsq[c] * inv - mu * mu;
  const float sc  = gamma[c] * __frsqrt_rn(var + 1e-5f);
  scale[c] = sc;
  shift[c] = beta[c] - mu * sc;
}

__global__ __launch_bounds__(256) void bn_relu_to_f16(
    const float* __restrict__ y, const float* __restrict__ scale,
    const float* __restrict__ shift, _Float16* __restrict__ o, int total) {
  int i = blockIdx.x * 256 + threadIdx.x;
  if (i >= total) return;
  const int c = i & 63;
  float v = y[i] * scale[c] + shift[c];
  v = v > 0.0f ? v : 0.0f;
  o[i] = (_Float16)v;
}

__global__ __launch_bounds__(256) void bn_res_relu(
    const float* __restrict__ y, const float* __restrict__ scale,
    const float* __restrict__ shift, const float* __restrict__ feats,
    float* __restrict__ out, int total) {
  int i = blockIdx.x * 256 + threadIdx.x;
  if (i >= total) return;
  const int c = i & 63;
  const float v = y[i] * scale[c] + shift[c] + feats[i];
  out[i] = v > 0.0f ? v : 0.0f;
}

// ------------------------------- launcher ----------------------------------

extern "C" void kernel_launch(void* const* d_in, const int* in_sizes, int n_in,
                              void* d_out, int out_size, void* d_ws, size_t ws_size,
                              hipStream_t stream) {
  const float* feats = (const float*)d_in[0];
  const int*   nbr   = (const int*)d_in[1];
  const float* W1    = (const float*)d_in[2];
  const float* b1    = (const float*)d_in[3];
  const float* g1    = (const float*)d_in[4];
  const float* be1   = (const float*)d_in[5];
  const float* W2    = (const float*)d_in[6];
  const float* b2    = (const float*)d_in[7];
  const float* g2    = (const float*)d_in[8];
  const float* be2   = (const float*)d_in[9];
  float* out = (float*)d_out;

  const int nelem = NVOX * CCH;          // 6,400,000
  const int wtot  = KOFF * CCH * CCH;    // 110,592

  // Workspace carve-up (all 256B aligned).
  char* ws = (char*)d_ws;
  _Float16* bufH = (_Float16*)ws;                         // 12,800,000 B (feats_h, then h)
  float*    y    = (float*)(ws + 12800000);               // 25,600,000 B (y1, then y2)
  _Float16* W1t  = (_Float16*)(ws + 38400000);            //    221,184 B
  _Float16* W2t  = (_Float16*)(ws + 38621184);            //    221,184 B
  float*    stats = (float*)(ws + 38842368);              //      1,024 B
  float* sum1 = stats;       float* sq1 = stats + 64;
  float* sum2 = stats + 128; float* sq2 = stats + 192;
  float* scsh = (float*)(ws + 38843392);                  //      1,024 B
  float* sc1 = scsh;        float* sh1 = scsh + 64;
  float* sc2 = scsh + 128;  float* sh2 = scsh + 192;

  // 0) zero BN accumulators (graph-capturable memset node)
  hipMemsetAsync(stats, 0, 256 * sizeof(float), stream);

  // 1) feats -> f16
  cvt_f32_to_f16<<<(nelem + 255) / 256, 256, 0, stream>>>(feats, bufH, nelem);
  // 2) weights -> transposed f16
  transpose_w<<<(wtot + 255) / 256, 256, 0, stream>>>(W1, W1t, wtot);
  transpose_w<<<(wtot + 255) / 256, 256, 0, stream>>>(W2, W2t, wtot);

  const int convBlocks = (NTILES + 7) / 8;   // 8 waves (tiles) per 256-thread block
  // 3) conv1
  conv_wmma<<<convBlocks, 256, LDSBYTES, stream>>>(bufH, nbr, W1t, b1, y, sum1, sq1);
  // 4) BN1 fold
  bn_stats<<<1, 64, 0, stream>>>(sum1, sq1, g1, be1, sc1, sh1);
  // 5) h = relu(bn1(y1)) -> f16 (reuses bufH)
  bn_relu_to_f16<<<(nelem + 255) / 256, 256, 0, stream>>>(y, sc1, sh1, bufH, nelem);
  // 6) conv2 (reuses y)
  conv_wmma<<<convBlocks, 256, LDSBYTES, stream>>>(bufH, nbr, W2t, b2, y, sum2, sq2);
  // 7) BN2 fold
  bn_stats<<<1, 64, 0, stream>>>(sum2, sq2, g2, be2, sc2, sh2);
  // 8) out = relu(bn2(y2) + feats)
  bn_res_relu<<<(nelem + 255) / 256, 256, 0, stream>>>(y, sc2, sh2, feats, out, nelem);
}
